// Phase56Processor_51548197487050
// MI455X (gfx1250) — matrix operand order repaired and checked by reading, not compiled, not run on hardware
//
#include <hip/hip_runtime.h>

typedef __bf16 bf16_t;
typedef __attribute__((ext_vector_type(16))) __bf16 v16bf;
typedef __attribute__((ext_vector_type(8)))  __bf16 v8bf;
typedef __attribute__((ext_vector_type(8)))  float  v8f;

// ---------- WMMA helpers (CDNA5 wave32, 16x16x32 bf16, f32 accum) ----------

__device__ __forceinline__ v8f vzero8() {
    v8f z;
#pragma unroll
    for (int i = 0; i < 8; ++i) z[i] = 0.0f;
    return z;
}

// A fragment: 16x32 bf16 tile. Lane l: row M = l&15; elements 0..15 = K0..K0+15,
// K0 = (l<16)?0:16.  A is row-major, pre-offset to tile row.
__device__ __forceinline__ v16bf load_a_frag(const bf16_t* A, int lda, int kk, int lane) {
    int m  = lane & 15;
    int k0 = kk + (lane & 16);                 // +16 for upper half-wave
    const bf16_t* p = A + (size_t)m * lda + k0;
    v8bf lo = *(const v8bf*)(p);
    v8bf hi = *(const v8bf*)(p + 8);
    return __builtin_shufflevector(lo, hi, 0,1,2,3,4,5,6,7,8,9,10,11,12,13,14,15);
}

// B fragment: 32x16 bf16 tile, supplied as Bt[n][k] row-major (pre-offset to row n0).
// Lane l: col N = l&15; elements 0..15 = K = K0..K0+15, K0 = (l<16)?0:16.
__device__ __forceinline__ v16bf load_b_frag(const bf16_t* Bt, int ldb, int kk, int lane) {
    int n  = lane & 15;
    int k0 = kk + (lane & 16);                 // +16 for upper half-wave
    return *(const v16bf*)(Bt + (size_t)n * ldb + k0);
}

__device__ __forceinline__ v8f wmma_bf16(v16bf a, v16bf b, v8f c) {
    return __builtin_amdgcn_wmma_f32_16x16x32_bf16(false, a, false, b, (short)0, c, false, false);
}

// C store (bf16): lane l holds N=l&15, rows M = r + 8*(l>=16). C pre-offset to tile origin.
__device__ __forceinline__ void store_c_bf16(bf16_t* C, int ldc, const v8f& acc, int lane) {
    int n  = lane & 15;
    int mh = (lane & 16) >> 1;
#pragma unroll
    for (int r = 0; r < 8; ++r) C[(size_t)(r + mh) * ldc + n] = (bf16_t)acc[r];
}

// Register-blocked 32x64 tile: 2 m-subtiles x 4 n-subtiles, A-frags reused 4x, B-frags 2x.
__device__ __forceinline__ void gemm_block_2x4(const bf16_t* A0, const bf16_t* Bn,
                                               int K, int lda, int ldb, int lane,
                                               v8f acc[2][4]) {
    for (int kk = 0; kk < K; kk += 32) {
        v16bf a0 = load_a_frag(A0, lda, kk, lane);
        v16bf a1 = load_a_frag(A0 + (size_t)16 * lda, lda, kk, lane);
        v16bf b0 = load_b_frag(Bn + (size_t)0  * ldb, ldb, kk, lane);
        v16bf b1 = load_b_frag(Bn + (size_t)16 * ldb, ldb, kk, lane);
        v16bf b2 = load_b_frag(Bn + (size_t)32 * ldb, ldb, kk, lane);
        v16bf b3 = load_b_frag(Bn + (size_t)48 * ldb, ldb, kk, lane);
        acc[0][0] = wmma_bf16(a0, b0, acc[0][0]);
        acc[0][1] = wmma_bf16(a0, b1, acc[0][1]);
        acc[0][2] = wmma_bf16(a0, b2, acc[0][2]);
        acc[0][3] = wmma_bf16(a0, b3, acc[0][3]);
        acc[1][0] = wmma_bf16(a1, b0, acc[1][0]);
        acc[1][1] = wmma_bf16(a1, b1, acc[1][1]);
        acc[1][2] = wmma_bf16(a1, b2, acc[1][2]);
        acc[1][3] = wmma_bf16(a1, b3, acc[1][3]);
    }
}

// ---------- small prep kernels ----------

__global__ void cvt_f32_bf16_kernel(const float* __restrict__ in, bf16_t* __restrict__ out, size_t n) {
    size_t i = (size_t)blockIdx.x * blockDim.x + threadIdx.x;
    if (i < n) out[i] = (bf16_t)in[i];
}

__global__ void zero_bf16_kernel(bf16_t* __restrict__ p, size_t n) {
    size_t i = (size_t)blockIdx.x * blockDim.x + threadIdx.x;
    if (i < n) p[i] = (bf16_t)0.0f;
}

// in[rows][cols] (f32) -> out[cols][rows] (bf16)   (Bt layout builder)
__global__ void transpose_cvt_kernel(const float* __restrict__ in, bf16_t* __restrict__ out,
                                     int rows, int cols) {
    size_t i = (size_t)blockIdx.x * blockDim.x + threadIdx.x;
    if (i < (size_t)rows * cols) {
        int rr = (int)(i / cols), cc = (int)(i % cols);
        out[(size_t)cc * rows + rr] = (bf16_t)in[i];
    }
}

// Y[m][r] = sum_k X[m][k] * A[k][r]  (rank-4 LoRA pre-projection, f32 X)
__global__ void lora_proj_f32_kernel(const float* __restrict__ X, const float* __restrict__ A,
                                     float* __restrict__ Y, int M, int K) {
    int idx = blockIdx.x * blockDim.x + threadIdx.x;
    if (idx < M * 4) {
        int m = idx >> 2, r = idx & 3;
        float s = 0.0f;
        for (int k = 0; k < K; ++k) s += X[(size_t)m * K + k] * A[k * 4 + r];
        Y[idx] = s;
    }
}

// same but bf16 X
__global__ void lora_proj_bf16_kernel(const bf16_t* __restrict__ X, const float* __restrict__ A,
                                      float* __restrict__ Y, int M, int K) {
    int idx = blockIdx.x * blockDim.x + threadIdx.x;
    if (idx < M * 4) {
        int m = idx >> 2, r = idx & 3;
        float s = 0.0f;
        for (int k = 0; k < K; ++k) s += (float)X[(size_t)m * K + k] * A[k * 4 + r];
        Y[idx] = s;
    }
}

// ---------- K/V projection GEMM with LoRA epilogue ----------
// enc_bf [624x768] @ Wbt[n][k] -> [616 x 1280]; + encA[m][0:4] . loraB[0:4][n]
// is_v==0: Kout[b][h][t][64] ; is_v==1: Vtout[b][h][d][96] (transposed, t-padded)
__global__ void kv_gemm_kernel(const bf16_t* __restrict__ enc_bf, const bf16_t* __restrict__ Wbt,
                               const float* __restrict__ encA, const float* __restrict__ loraB,
                               bf16_t* __restrict__ Kout, bf16_t* __restrict__ Vtout, int is_v) {
    const int Cc = 768, Dd = 1280, Tt = 77, Hh = 20;
    int lane = threadIdx.x & 31, wave = threadIdx.x >> 5;
    int m0 = blockIdx.x * 16;
    int n0 = (blockIdx.y * 4 + wave) * 16;
    const bf16_t* At = enc_bf + (size_t)m0 * Cc;
    const bf16_t* Bn = Wbt + (size_t)n0 * Cc;
    v8f acc = vzero8();
    for (int kk = 0; kk < Cc; kk += 32)
        acc = wmma_bf16(load_a_frag(At, Cc, kk, lane), load_b_frag(Bn, Cc, kk, lane), acc);
    int n = n0 + (lane & 15);
    int h = n >> 6, d = n & 63;
    int mh = (lane & 16) >> 1;
    float lb0 = loraB[0 * Dd + n], lb1 = loraB[1 * Dd + n];
    float lb2 = loraB[2 * Dd + n], lb3 = loraB[3 * Dd + n];
#pragma unroll
    for (int r = 0; r < 8; ++r) {
        int m = m0 + r + mh;
        if (m < 8 * Tt) {
            int b = m / Tt, t = m - b * Tt;
            float v = acc[r] + encA[m * 4 + 0] * lb0 + encA[m * 4 + 1] * lb1
                            + encA[m * 4 + 2] * lb2 + encA[m * 4 + 3] * lb3;
            if (is_v) Vtout[((size_t)(b * Hh + h) * 64 + d) * 96 + t] = (bf16_t)v;
            else      Kout [((size_t)(b * Hh + h) * 80 + t) * 64 + d] = (bf16_t)v;
        }
    }
}

// ---------- register-blocked bf16 GEMM (Q projection): 32x64 per wave ----------
__global__ void gemm_bf16_blocked_kernel(const bf16_t* __restrict__ A, const bf16_t* __restrict__ Bt,
                                         bf16_t* __restrict__ C, int K, int lda, int ldb, int ldc) {
    int lane = threadIdx.x & 31, wave = threadIdx.x >> 5;
    size_t m0 = (size_t)blockIdx.x * 32;
    int n0 = (blockIdx.y * 4 + wave) * 64;
    v8f acc[2][4];
#pragma unroll
    for (int i = 0; i < 2; ++i)
#pragma unroll
        for (int j = 0; j < 4; ++j) acc[i][j] = vzero8();
    gemm_block_2x4(A + m0 * lda, Bt + (size_t)n0 * ldb, K, lda, ldb, lane, acc);
#pragma unroll
    for (int mi = 0; mi < 2; ++mi)
#pragma unroll
        for (int nt = 0; nt < 4; ++nt)
            store_c_bf16(C + (m0 + mi * 16) * ldc + n0 + nt * 16, ldc, acc[mi][nt], lane);
}

// ---------- fused attention: scores -> softmax -> PV ----------
__global__ void attn_kernel(const bf16_t* __restrict__ Q, const bf16_t* __restrict__ Kb,
                            const bf16_t* __restrict__ Vt, bf16_t* __restrict__ O) {
    const int Ss = 4096, Dd = 1280, Hh = 20;
    __shared__ bf16_t wlds[4][16][96];
    int lane = threadIdx.x & 31, wave = threadIdx.x >> 5;
    int bh = blockIdx.y, b = bh / Hh, h = bh % Hh;
    int s0 = blockIdx.x * 64 + wave * 16;
    const bf16_t* Qt = Q + ((size_t)b * Ss + s0) * Dd + h * 64;
    const bf16_t* Kt = Kb + (size_t)(b * Hh + h) * 80 * 64;

    v8f sc[5];
#pragma unroll
    for (int i = 0; i < 5; ++i) sc[i] = vzero8();
#pragma unroll
    for (int kk = 0; kk < 64; kk += 32) {
        v16bf a = load_a_frag(Qt, Dd, kk, lane);
#pragma unroll
        for (int nt = 0; nt < 5; ++nt)
            sc[nt] = wmma_bf16(a, load_b_frag(Kt + (size_t)nt * 16 * 64, 64, kk, lane), sc[nt]);
    }

    const float scale = 0.125f;   // dh=64 -> 1/sqrt(64)
    int nl = lane & 15, mh = (lane & 16) >> 1;
#pragma unroll
    for (int r = 0; r < 8; ++r) {
        float vals[5], mx = -1e30f;
#pragma unroll
        for (int nt = 0; nt < 5; ++nt) {
            int t = nt * 16 + nl;
            float v = (t < 77) ? sc[nt][r] * scale : -1e30f;
            vals[nt] = v;
            mx = fmaxf(mx, v);
        }
        for (int off = 1; off < 16; off <<= 1) mx = fmaxf(mx, __shfl_xor(mx, off, 32));
        float sum = 0.0f;
#pragma unroll
        for (int nt = 0; nt < 5; ++nt) { float e = __expf(vals[nt] - mx); vals[nt] = e; sum += e; }
        for (int off = 1; off < 16; off <<= 1) sum += __shfl_xor(sum, off, 32);
        float inv = 1.0f / sum;
        int M = r + mh;
#pragma unroll
        for (int nt = 0; nt < 5; ++nt) wlds[wave][M][nt * 16 + nl] = (bf16_t)(vals[nt] * inv);
    }
    // zero-pad columns 80..95 so the padded K-steps contribute nothing
#pragma unroll
    for (int r = 0; r < 8; ++r) wlds[wave][r + mh][80 + nl] = (bf16_t)0.0f;
    __syncthreads();

    const bf16_t* wA = &wlds[wave][0][0];
    const bf16_t* Vb = Vt + (size_t)(b * Hh + h) * 64 * 96;
    bf16_t* Ot = O + ((size_t)b * Ss + s0) * Dd + h * 64;
    // hoist the 3 invariant A-fragments (w tile) out of the n-tile loop
    v16bf wa0 = load_a_frag(wA, 96, 0,  lane);
    v16bf wa1 = load_a_frag(wA, 96, 32, lane);
    v16bf wa2 = load_a_frag(wA, 96, 64, lane);
#pragma unroll
    for (int nt = 0; nt < 4; ++nt) {
        const bf16_t* Vn = Vb + (size_t)nt * 16 * 96;
        v8f acc = vzero8();
        acc = wmma_bf16(wa0, load_b_frag(Vn, 96, 0,  lane), acc);
        acc = wmma_bf16(wa1, load_b_frag(Vn, 96, 32, lane), acc);
        acc = wmma_bf16(wa2, load_b_frag(Vn, 96, 64, lane), acc);
        store_c_bf16(Ot + nt * 16, Dd, acc, lane);
    }
}

// ---------- fused residual adapter: out2 = O + relu(O @ down) @ up ----------
__global__ void adapter_kernel(const bf16_t* __restrict__ O, const bf16_t* __restrict__ down_bt,
                               const bf16_t* __restrict__ up_bt, bf16_t* __restrict__ out2) {
    const int Dd = 1280;
    __shared__ bf16_t mlds[4][16][64];
    int lane = threadIdx.x & 31, wave = threadIdx.x >> 5;
    size_t m0 = ((size_t)blockIdx.x * 4 + wave) * 16;
    const bf16_t* At = O + m0 * Dd;

    v8f acc[4];
#pragma unroll
    for (int i = 0; i < 4; ++i) acc[i] = vzero8();
    for (int kk = 0; kk < Dd; kk += 32) {
        v16bf a = load_a_frag(At, Dd, kk, lane);
#pragma unroll
        for (int nt = 0; nt < 4; ++nt)
            acc[nt] = wmma_bf16(a, load_b_frag(down_bt + (size_t)nt * 16 * Dd, Dd, kk, lane), acc[nt]);
    }
    int nl = lane & 15, mh = (lane & 16) >> 1;
#pragma unroll
    for (int nt = 0; nt < 4; ++nt)
#pragma unroll
        for (int r = 0; r < 8; ++r)
            mlds[wave][r + mh][nt * 16 + nl] = (bf16_t)fmaxf(acc[nt][r], 0.0f);
    __syncthreads();

    // hoist the 2 invariant A-fragments (relu(mid) tile) out of the 80-iteration loop
    const bf16_t* wA = &mlds[wave][0][0];
    v16bf ma0 = load_a_frag(wA, 64, 0,  lane);
    v16bf ma1 = load_a_frag(wA, 64, 32, lane);
    for (int nt = 0; nt < 80; ++nt) {
        const bf16_t* Un = up_bt + (size_t)nt * 16 * 64;
        v8f a2 = vzero8();
        a2 = wmma_bf16(ma0, load_b_frag(Un, 64, 0,  lane), a2);
        a2 = wmma_bf16(ma1, load_b_frag(Un, 64, 32, lane), a2);
        int n = nt * 16 + nl;
#pragma unroll
        for (int r = 0; r < 8; ++r) {
            size_t m = m0 + r + mh;
            out2[m * Dd + n] = (bf16_t)((float)O[m * Dd + n] + a2[r]);
        }
    }
}

// ---------- output projection, register-blocked, bias + rank-4 LoRA epilogue (f32 out) ----------
__global__ void outproj_blocked_kernel(const bf16_t* __restrict__ out2, const bf16_t* __restrict__ Wout_bt,
                                       const float* __restrict__ bias, const float* __restrict__ lora_vec,
                                       const float* __restrict__ loraB, float* __restrict__ R) {
    const int Dd = 1280;
    int lane = threadIdx.x & 31, wave = threadIdx.x >> 5;
    size_t m0 = (size_t)blockIdx.x * 32;
    int n0 = (blockIdx.y * 4 + wave) * 64;
    v8f acc[2][4];
#pragma unroll
    for (int i = 0; i < 2; ++i)
#pragma unroll
        for (int j = 0; j < 4; ++j) acc[i][j] = vzero8();
    gemm_block_2x4(out2 + m0 * Dd, Wout_bt + (size_t)n0 * Dd, Dd, Dd, Dd, lane, acc);

    int nl = lane & 15, mh = (lane & 16) >> 1;
#pragma unroll
    for (int mi = 0; mi < 2; ++mi) {
        // per-row LoRA activations (shared across the 4 n-subtiles)
        float lv[8][4];
#pragma unroll
        for (int r = 0; r < 8; ++r) {
            size_t m = m0 + mi * 16 + r + mh;
#pragma unroll
            for (int j = 0; j < 4; ++j) lv[r][j] = lora_vec[m * 4 + j];
        }
#pragma unroll
        for (int nt = 0; nt < 4; ++nt) {
            int n = n0 + nt * 16 + nl;
            float bb = bias[n];
            float l0 = loraB[n], l1 = loraB[Dd + n], l2 = loraB[2 * Dd + n], l3 = loraB[3 * Dd + n];
#pragma unroll
            for (int r = 0; r < 8; ++r) {
                size_t m = m0 + mi * 16 + r + mh;
                R[m * Dd + n] = acc[mi][nt][r] + bb + lv[r][0] * l0 + lv[r][1] * l1
                                               + lv[r][2] * l2 + lv[r][3] * l3;
            }
        }
    }
}

// ---------- host launch ----------

extern "C" void kernel_launch(void* const* d_in, const int* in_sizes, int n_in,
                              void* d_out, int out_size, void* d_ws, size_t ws_size,
                              hipStream_t stream) {
    const int B = 8, S = 4096, T = 77, D = 1280, C = 768, H = 20;
    const float* hidden = (const float*)d_in[0];
    const float* enc    = (const float*)d_in[1];
    const float* Wq     = (const float*)d_in[2];
    const float* Wk     = (const float*)d_in[3];
    const float* Wv     = (const float*)d_in[4];
    const float* Wout   = (const float*)d_in[5];
    const float* b_out  = (const float*)d_in[6];
    const float* lora_k_A  = (const float*)d_in[7];
    const float* lora_k_B  = (const float*)d_in[8];
    const float* lora_v_A  = (const float*)d_in[9];
    const float* lora_v_B  = (const float*)d_in[10];
    const float* lora_o_A  = (const float*)d_in[11];
    const float* lora_o_B  = (const float*)d_in[12];
    const float* adapter_down = (const float*)d_in[13];
    const float* adapter_up   = (const float*)d_in[14];
    float* result = (float*)d_out;

    char* w = (char*)d_ws;
    auto alloc = [&](size_t bytes) -> char* {
        char* p = w; w += (bytes + 255) & ~(size_t)255; return p;
    };
    const size_t MS = (size_t)B * S;              // 32768
    bf16_t* hidden_bf = (bf16_t*)alloc(MS * D * 2);   // region R1, reused as O
    bf16_t* Qbuf      = (bf16_t*)alloc(MS * D * 2);   // region R2, reused as out2
    bf16_t* enc_bf    = (bf16_t*)alloc((size_t)640 * C * 2);  // 616 rows padded to 640
    bf16_t* Kbuf      = (bf16_t*)alloc((size_t)B * H * 80 * 64 * 2);
    bf16_t* Vtbuf     = (bf16_t*)alloc((size_t)B * H * 64 * 96 * 2);
    bf16_t* Wq_bt     = (bf16_t*)alloc((size_t)D * D * 2);
    bf16_t* Wk_bt     = (bf16_t*)alloc((size_t)D * C * 2);
    bf16_t* Wv_bt     = (bf16_t*)alloc((size_t)D * C * 2);
    bf16_t* Wout_bt   = (bf16_t*)alloc((size_t)D * D * 2);
    bf16_t* down_bt   = (bf16_t*)alloc((size_t)64 * D * 2);
    bf16_t* up_bt     = (bf16_t*)alloc((size_t)D * 64 * 2);
    float*  encA_k    = (float*)alloc((size_t)616 * 4 * 4);
    float*  encA_v    = (float*)alloc((size_t)616 * 4 * 4);
    float*  lora_vec  = (float*)alloc(MS * 4 * 4);
    bf16_t* Obuf = hidden_bf;   // alias: hidden_bf dead after Q GEMM
    bf16_t* out2 = Qbuf;        // alias: Q dead after attention

    auto gsz = [](size_t n, int t) { return (unsigned)((n + t - 1) / t); };

    // prep: zero pads, converts, weight layout, LoRA pre-projections
    zero_bf16_kernel<<<gsz((size_t)640 * C, 256), 256, 0, stream>>>(enc_bf, (size_t)640 * C);
    zero_bf16_kernel<<<gsz((size_t)B * H * 64 * 96, 256), 256, 0, stream>>>(Vtbuf, (size_t)B * H * 64 * 96);
    cvt_f32_bf16_kernel<<<gsz(MS * D, 256), 256, 0, stream>>>(hidden, hidden_bf, MS * D);
    cvt_f32_bf16_kernel<<<gsz((size_t)616 * C, 256), 256, 0, stream>>>(enc, enc_bf, (size_t)616 * C);
    cvt_f32_bf16_kernel<<<gsz((size_t)D * D, 256), 256, 0, stream>>>(Wq, Wq_bt, (size_t)D * D);
    cvt_f32_bf16_kernel<<<gsz((size_t)D * C, 256), 256, 0, stream>>>(Wk, Wk_bt, (size_t)D * C);
    cvt_f32_bf16_kernel<<<gsz((size_t)D * C, 256), 256, 0, stream>>>(Wv, Wv_bt, (size_t)D * C);
    cvt_f32_bf16_kernel<<<gsz((size_t)D * D, 256), 256, 0, stream>>>(Wout, Wout_bt, (size_t)D * D);
    transpose_cvt_kernel<<<gsz((size_t)D * 64, 256), 256, 0, stream>>>(adapter_down, down_bt, D, 64);
    transpose_cvt_kernel<<<gsz((size_t)64 * D, 256), 256, 0, stream>>>(adapter_up, up_bt, 64, D);
    lora_proj_f32_kernel<<<gsz((size_t)616 * 4, 256), 256, 0, stream>>>(enc, lora_k_A, encA_k, 616, C);
    lora_proj_f32_kernel<<<gsz((size_t)616 * 4, 256), 256, 0, stream>>>(enc, lora_v_A, encA_v, 616, C);

    // K and V projections (M=624 tiles, N=1280, K=768) with LoRA epilogue
    kv_gemm_kernel<<<dim3(39, 20), 128, 0, stream>>>(enc_bf, Wk_bt, encA_k, lora_k_B, Kbuf, Vtbuf, 0);
    kv_gemm_kernel<<<dim3(39, 20), 128, 0, stream>>>(enc_bf, Wv_bt, encA_v, lora_v_B, Kbuf, Vtbuf, 1);

    // Q projection (32768 x 1280 x 1280), 32x64 tiles per wave
    gemm_bf16_blocked_kernel<<<dim3((unsigned)(MS / 32), 5), 128, 0, stream>>>(
        hidden_bf, Wq_bt, Qbuf, D, D, D, D);

    // fused attention
    attn_kernel<<<dim3(S / 64, B * H), 128, 0, stream>>>(Qbuf, Kbuf, Vtbuf, Obuf);

    // fused residual adapter
    adapter_kernel<<<gsz(MS / 16, 4), 128, 0, stream>>>(Obuf, down_bt, up_bt, out2);

    // rank-4 LoRA pre-projection on out2, then output projection (32x64 tiles per wave)
    lora_proj_bf16_kernel<<<gsz(MS * 4, 256), 256, 0, stream>>>(out2, lora_o_A, lora_vec, (int)MS, D);
    outproj_blocked_kernel<<<dim3((unsigned)(MS / 32), 5), 128, 0, stream>>>(
        out2, Wout_bt, b_out, lora_vec, lora_o_B, result);
}
